// OptPosEnc_51281909514403
// MI455X (gfx1250) — matrix operand order repaired and checked by reading, not compiled, not run on hardware
//
#include <hip/hip_runtime.h>

// ---------------------------------------------------------------------------
// OptPosEnc on MI455X (gfx1250, wave32)
//
// out(P x 64) = S(P x 192) x shape_code^T(192 x 64), S sparse (6 nnz/row).
// Store-bandwidth bound: ~134 MB traffic -> ~5.8 us floor @ 23.3 TB/s.
// Dense contraction on v_wmma_f32_16x16x32_f16:
//   - codebook held in 192 VGPRs/wave, reused across ~15 tiles
//   - sparse A staged in LDS: incremental zero (clear only last tile's
//     cells) + f16 scatter; A-fragments read as 2x ds_load_b128 per chunk,
//     one chunk prefetched ahead of the 4 WMMAs consuming the previous one
//   - D stored as 32x global_store_b32 with immediate offsets from ONE base
//     (no cross-acc register packing -> no WMMA->VALU hazard NOPs)
// ---------------------------------------------------------------------------

typedef __attribute__((ext_vector_type(16))) _Float16 v16h;
typedef __attribute__((ext_vector_type(8)))  _Float16 v8h;
typedef __attribute__((ext_vector_type(8)))  float    v8f;

#define PT_NUM   500000
#define NCH      64
#define KTOT     192            // 3 axes * 64 codes
#define NTILES   (PT_NUM / 16)  // 31250, exact
#define ROWH     200            // padded LDS row stride in halfs (bank spread)
#define WVB      4              // waves per block

__global__ __launch_bounds__(128, 1)
void posenc_wmma(const float* __restrict__ coords,
                 const float* __restrict__ sc,
                 float* __restrict__ out)
{
    __shared__ __align__(16) _Float16 Abuf[WVB][16][ROWH];  // 25.6 KB

    const int tid  = threadIdx.x;
    const int lane = tid & 31;
    const int wv   = tid >> 5;
    const int hi   = lane >> 4;   // 0: lanes 0-15, 1: lanes 16-31
    const int ml   = lane & 15;   // A row (point) / B-D column within tile

    // ---- Load B = shape_code^T into registers, f32 -> f16 -----------------
    // N-tile t covers channels [16t, 16t+16): column n = channel 16t + n.
    // B(32x16) in this code's contiguous K order: lanes 0-15 hold K=0..15, lanes 16-31
    // K=16..31 -> per lane 16 consecutive K values (4x b128 global loads).
    v16h B[6][4];
    #pragma unroll
    for (int kc = 0; kc < 6; ++kc) {
        #pragma unroll
        for (int t = 0; t < 4; ++t) {
            const int ch = t * 16 + ml;                    // physical channel
            const float* src = sc + ch * KTOT + kc * 32 + hi * 16;
            #pragma unroll
            for (int s = 0; s < 16; ++s)
                B[kc][t][s] = (_Float16)src[s];
        }
    }

    // ---- One-time full clear of this wave's A staging ---------------------
    {
        v8h z = {};
        _Float16* rowz = &Abuf[wv][ml][hi * 96];
        #pragma unroll
        for (int i = 0; i < 12; ++i)
            *(v8h*)(rowz + i * 8) = z;                     // 12x ds_store_b128
    }
    __builtin_amdgcn_wave_barrier();

    // scalar (SGPR) tile loop bookkeeping
    const int waveId = __builtin_amdgcn_readfirstlane(blockIdx.x * WVB + wv);
    const int nWaves = gridDim.x * WVB;

    // previous-tile scatter positions (incremental zeroing)
    int pgx = 0, pgy = 64, pgz = 128;

    // A-fragment loader: lane (hi,ml) holds the run
    // K = kc*32 + hi*16 + {0..15} (matches B): 2x ds_load_b128.
    auto loadA = [&](int kc) -> v16h {
        const _Float16* row = &Abuf[wv][ml][kc * 32 + hi * 16];
        const v8h alo = *(const v8h*)row;
        const v8h ahi = *(const v8h*)(row + 8);
        v16h a;
        #pragma unroll
        for (int i = 0; i < 8; ++i) { a[i] = alo[i]; a[8 + i] = ahi[i]; }
        return a;
    };

    for (int tile = waveId; tile < NTILES; tile += nWaves) {
        const int p0 = tile * 16;

        // ---- coords for point p0+ml (all lanes; hi halves split the axes)
        const float* cp = coords + (unsigned)(p0 + ml) * 3u;
        const float cx = (cp[0] + 1.0f) * 31.5f;
        const float cy = (cp[1] + 1.0f) * 31.5f;
        const float cz = (cp[2] + 1.0f) * 31.5f;

        // ---- clear last tile's cells, then scatter this tile's weights ---
        // LDS pipe is in-order within a wave; no dscnt drain needed between
        // the scatter stores and the fragment loads.
        if (hi == 0) {          // lanes 0-15: axes 0 and 1
            Abuf[wv][ml][pgx] = (_Float16)0.f; Abuf[wv][ml][pgx + 1] = (_Float16)0.f;
            Abuf[wv][ml][pgy] = (_Float16)0.f; Abuf[wv][ml][pgy + 1] = (_Float16)0.f;
            const float fx = floorf(cx), fy = floorf(cy);
            const float wx = cx - fx,    wy = cy - fy;
            const int gx = (int)fx;          // axis0 cols [0,64)
            const int gy = 64 + (int)fy;     // axis1 cols [64,128)
            Abuf[wv][ml][gx]     = (_Float16)(1.0f - wx);
            Abuf[wv][ml][gx + 1] = (_Float16)wx;
            Abuf[wv][ml][gy]     = (_Float16)(1.0f - wy);
            Abuf[wv][ml][gy + 1] = (_Float16)wy;
            pgx = gx; pgy = gy;
        } else {                // lanes 16-31: axis 2
            Abuf[wv][ml][pgz] = (_Float16)0.f; Abuf[wv][ml][pgz + 1] = (_Float16)0.f;
            const float fz = floorf(cz);
            const float wz = cz - fz;
            const int gz = 128 + (int)fz;    // axis2 cols [128,192)
            Abuf[wv][ml][gz]     = (_Float16)(1.0f - wz);
            Abuf[wv][ml][gz + 1] = (_Float16)wz;
            pgz = gz;
        }
        __builtin_amdgcn_wave_barrier();

        // ---- 6 K-chunks: D(16x16) += A(16x32) x B(32x16), 4 N-tiles ------
        // one-chunk-ahead A prefetch keeps ds_load latency off the WMMA path
        v8f acc0 = {}, acc1 = {}, acc2 = {}, acc3 = {};
        v16h aCur = loadA(0);
        #pragma unroll
        for (int kc = 0; kc < 6; ++kc) {
            const v16h aNxt = (kc < 5) ? loadA(kc + 1) : aCur;
            acc0 = __builtin_amdgcn_wmma_f32_16x16x32_f16(
                       false, aCur, false, B[kc][0], (short)0, acc0, false, false);
            acc1 = __builtin_amdgcn_wmma_f32_16x16x32_f16(
                       false, aCur, false, B[kc][1], (short)0, acc1, false, false);
            acc2 = __builtin_amdgcn_wmma_f32_16x16x32_f16(
                       false, aCur, false, B[kc][2], (short)0, acc2, false, false);
            acc3 = __builtin_amdgcn_wmma_f32_16x16x32_f16(
                       false, aCur, false, B[kc][3], (short)0, acc3, false, false);
            aCur = aNxt;
        }

        // ---- Store D: f32 C/D layout: VGPR r -> row r + 8*hi, col ml -----
        // One 32-bit base; every store is base + immediate (t*64 + r*256 B).
        float* obase = out + ((unsigned)(p0 + 8 * hi) * (unsigned)NCH + (unsigned)ml);
        #pragma unroll
        for (int r = 0; r < 8; ++r) {
            obase[(unsigned)r * NCH]      = acc0[r];
            obase[(unsigned)r * NCH + 16] = acc1[r];
            obase[(unsigned)r * NCH + 32] = acc2[r];
            obase[(unsigned)r * NCH + 48] = acc3[r];
        }
        // no tail barrier: lets next tile's coords load / clear hoist into
        // the WMMA + store shadow; LDS in-wave ordering keeps A consistent
    }
}

extern "C" void kernel_launch(void* const* d_in, const int* in_sizes, int n_in,
                              void* d_out, int out_size, void* d_ws, size_t ws_size,
                              hipStream_t stream)
{
    (void)in_sizes; (void)n_in; (void)d_ws; (void)ws_size; (void)out_size;
    const float* coords = (const float*)d_in[0];  // (1, 500000, 3) f32
    const float* sc     = (const float*)d_in[1];  // (64, 192) f32
    float*       out    = (float*)d_out;          // (1, 500000, 64) f32

    // 512 blocks x 4 waves = 2048 waves; ~15 tiles/wave amortizes the
    // one-time codebook->VGPR load while keeping multiple waves resident
    // per SIMD for DS/WMMA/store latency overlap.
    dim3 grid(512), block(128);
    hipLaunchKernelGGL(posenc_wmma, grid, block, 0, stream, coords, sc, out);
}
